// MassivePool_6150393168359
// MI455X (gfx1250) — matrix operand order repaired and checked by reading, not compiled, not run on hardware
//
#include <hip/hip_runtime.h>
#include <hip/hip_bf16.h>
#include <math.h>

// ---------------------------------------------------------------------------
// MassivePool retrieval pipeline for MI455X (gfx1250, wave32, WMMA).
//   B=4, S=1024 (BS=4096 rows), Dr=512, P=65536, Dp=1024, TOP_K=32
// Phase 0: f32 -> bf16 conversions (query, keys, W_out)
// Phase 1: fused scores GEMM (bf16 WMMA, double-buffered B) + streaming
//          top-32 per row, P split 4-ways for occupancy
// Phase 1b: merge the 4 partial top-32 lists per row
// Phase 2: softmax(32) + pool gather-aggregate -> bf16 aggregated
// Phase 3: output projection GEMM (bf16 WMMA, f32 accumulate) -> d_out
// ---------------------------------------------------------------------------

typedef __attribute__((ext_vector_type(8)))  float   v8f;
typedef __attribute__((ext_vector_type(16))) __bf16  v16bf;
typedef __attribute__((ext_vector_type(8)))  __bf16  v8bf;

#define BS_ROWS 4096
#define DR      512
#define POOL_P  65536
#define DP      1024
#define TOPK    32
#define NSPLIT  4
#define PSPLIT  (POOL_P / NSPLIT)   // 16384 columns per split

__device__ __forceinline__ unsigned short f32_to_bf16_bits(float f) {
    union { float f; unsigned int u; } x;
    x.f = f;
    unsigned int u = x.u;
    u += 0x7FFFu + ((u >> 16) & 1u);   // round-to-nearest-even
    return (unsigned short)(u >> 16);
}

// A fragment (16x32 bf16): per-lane one contiguous 16-elem K chunk.
__device__ __forceinline__ v16bf load_a_frag(const unsigned short* rowptr, int k, int lhalf) {
    return *(const v16bf*)(rowptr + k + lhalf * 16);
}

// B fragment (32x16 bf16): per-lane one contiguous 16-elem K chunk.
__device__ __forceinline__ v16bf load_b_frag(const unsigned short* rowptr, int k, int lhalf) {
    return *(const v16bf*)(rowptr + k + lhalf * 16);
}

// ---------------------------------------------------------------------------
// Phase 0: elementwise f32 -> bf16 (grid-stride)
// ---------------------------------------------------------------------------
__global__ void cvt_f32_bf16_kernel(const float* __restrict__ src,
                                    unsigned short* __restrict__ dst, int n) {
    for (int i = blockIdx.x * blockDim.x + threadIdx.x; i < n;
         i += gridDim.x * blockDim.x) {
        dst[i] = f32_to_bf16_bits(src[i]);
    }
}

// ---------------------------------------------------------------------------
// Phase 1: fused scores GEMM + streaming top-32 over one P-split.
// Grid: (128 M-tiles of 32 rows, 4 P-splits). Block: 128 threads (4 wave32).
// Wave w: rows (w>>1)*16..+15, columns (w&1)*32..+31 of a 64-wide N chunk
// (two 16x16 C accumulators per wave -> B double-buffer amortizes waits).
// ---------------------------------------------------------------------------
__global__ void __launch_bounds__(128)
scores_topk_kernel(const unsigned short* __restrict__ qb,   // [4096][512] bf16
                   const unsigned short* __restrict__ kb,   // [65536][512] bf16
                   float* __restrict__ topk_s_part,         // [NSPLIT][4096][32]
                   int*   __restrict__ topk_i_part) {       // [NSPLIT][4096][32]
    const int wgM   = blockIdx.x;          // 0..127
    const int split = blockIdx.y;          // 0..3
    const int tid   = threadIdx.x;         // 0..127
    const int wave  = tid >> 5;
    const int lane  = tid & 31;
    const int rowHalf = wave >> 1;         // which 16-row slab
    const int nsub    = wave & 1;          // which 32-col half of the 64 chunk
    const int lhalf   = lane >> 4;
    const int lrow    = lane & 15;
    const int mbase   = wgM * 32 + rowHalf * 16;
    const int nstart  = split * PSPLIT;
    const int nend    = nstart + PSPLIT;

    __shared__ float s_chunk[32][65];     // scores chunk (padded)
    __shared__ float s_ts[32][TOPK];      // running top-k scores
    __shared__ int   s_ti[32][TOPK];      // running top-k indices
    __shared__ float s_ps[32][64];        // pending candidates (scores)
    __shared__ int   s_pi[32][64];        // pending candidates (indices)
    __shared__ float s_min[32];           // current min of top-k per row
    __shared__ int   s_minslot[32];
    __shared__ int   s_pc[32];            // pending count per row

    for (int i = tid; i < 32 * TOPK; i += 128) {
        s_ts[i >> 5][i & 31] = -INFINITY;
        s_ti[i >> 5][i & 31] = 0;
    }
    if (tid < 32) {
        s_min[tid] = -INFINITY;
        s_minslot[tid] = 0;
        s_pc[tid] = 0;
    }
    __syncthreads();

    // A fragments (loop-invariant): 16 K-steps of the 16x512 row tile.
    const unsigned short* arow = qb + (size_t)(mbase + lrow) * DR;
    v16bf afrag[16];
#pragma unroll
    for (int kk = 0; kk < 16; ++kk)
        afrag[kk] = load_a_frag(arow, kk * 32, lhalf);

    // Chunk-level + K-level software pipeline of B fragments.
    const unsigned short* brow0 = kb + (size_t)(nstart + nsub * 32 + lrow) * DR;
    const unsigned short* brow1 = brow0 + (size_t)16 * DR;
    v16bf b0 = load_b_frag(brow0, 0, lhalf);
    v16bf b1 = load_b_frag(brow1, 0, lhalf);

    for (int nbase = nstart; nbase < nend; nbase += 64) {
        const int next_nbase = (nbase + 64 < nend) ? (nbase + 64) : nstart;
        const unsigned short* nbrow0 =
            kb + (size_t)(next_nbase + nsub * 32 + lrow) * DR;
        const unsigned short* nbrow1 = nbrow0 + (size_t)16 * DR;

        v8f c0 = {0.f, 0.f, 0.f, 0.f, 0.f, 0.f, 0.f, 0.f};
        v8f c1 = {0.f, 0.f, 0.f, 0.f, 0.f, 0.f, 0.f, 0.f};
#pragma unroll
        for (int kk = 0; kk < 16; ++kk) {
            v16bf t0, t1;
            if (kk < 15) {
                t0 = load_b_frag(brow0, (kk + 1) * 32, lhalf);
                t1 = load_b_frag(brow1, (kk + 1) * 32, lhalf);
            } else {      // prefetch first fragments of the NEXT chunk
                t0 = load_b_frag(nbrow0, 0, lhalf);
                t1 = load_b_frag(nbrow1, 0, lhalf);
            }
            c0 = __builtin_amdgcn_wmma_f32_16x16x32_bf16(
                     false, afrag[kk], false, b0, (short)0, c0, false, false);
            c1 = __builtin_amdgcn_wmma_f32_16x16x32_bf16(
                     false, afrag[kk], false, b1, (short)0, c1, false, false);
            b0 = t0;
            b1 = t1;
        }
        brow0 = nbrow0;
        brow1 = nbrow1;

        // Scatter both C tiles into the LDS chunk.
#pragma unroll
        for (int r = 0; r < 8; ++r) {
            const int lr = rowHalf * 16 + r + 8 * lhalf;
            s_chunk[lr][(nsub * 2 + 0) * 16 + lrow] = c0[r];
            s_chunk[lr][(nsub * 2 + 1) * 16 + lrow] = c1[r];
        }
        __syncthreads();

        // Parallel threshold filter: 4 threads per row x 16 candidates each.
        {
            const int frow = tid & 31;
            const int fsub = tid >> 5;
            const float thr = s_min[frow];
#pragma unroll
            for (int j = 0; j < 16; ++j) {
                const int ccol = fsub * 16 + j;
                const float sc = s_chunk[frow][ccol];
                if (sc > thr) {
                    const int p = atomicAdd(&s_pc[frow], 1);
                    s_ps[frow][p] = sc;
                    s_pi[frow][p] = nbase + ccol;
                }
            }
        }
        __syncthreads();

        // Serial insertion of (rare) surviving candidates, one thread/row.
        if (tid < 32) {
            const int row = tid;
            const int pc = s_pc[row];
            for (int j = 0; j < pc; ++j) {
                const float sc = s_ps[row][j];
                if (sc > s_min[row]) {
                    const int slot = s_minslot[row];
                    s_ts[row][slot] = sc;
                    s_ti[row][slot] = s_pi[row][j];
                    float m = s_ts[row][0];
                    int ms = 0;
                    for (int q = 1; q < TOPK; ++q) {
                        const float v = s_ts[row][q];
                        if (v < m) { m = v; ms = q; }
                    }
                    s_min[row] = m;
                    s_minslot[row] = ms;
                }
            }
            s_pc[row] = 0;
        }
        __syncthreads();
    }

    for (int i = tid; i < 32 * TOPK; i += 128) {
        const int row = i >> 5, slot = i & 31;
        const size_t grow = (size_t)split * BS_ROWS + (size_t)wgM * 32 + row;
        topk_s_part[grow * TOPK + slot] = s_ts[row][slot];
        topk_i_part[grow * TOPK + slot] = s_ti[row][slot];
    }
}

// ---------------------------------------------------------------------------
// Phase 1b: merge NSPLIT partial top-32 lists -> final top-32 per row.
// One block per row; 128 candidates; 32 argmax LDS reductions.
// ---------------------------------------------------------------------------
__global__ void __launch_bounds__(128)
merge_topk_kernel(const float* __restrict__ topk_s_part,
                  const int*   __restrict__ topk_i_part,
                  float* __restrict__ topk_s,
                  int*   __restrict__ topk_i) {
    const int row = blockIdx.x;     // 0..4095
    const int tid = threadIdx.x;    // 0..127  (NSPLIT*TOPK)

    __shared__ float sv[128];
    __shared__ int   si[128];
    __shared__ float rv[128];
    __shared__ int   ri[128];

    {
        const int split = tid >> 5, slot = tid & 31;
        const size_t base = ((size_t)split * BS_ROWS + row) * TOPK + slot;
        sv[tid] = topk_s_part[base];
        si[tid] = topk_i_part[base];
    }
    __syncthreads();

    for (int k = 0; k < TOPK; ++k) {
        rv[tid] = sv[tid];
        ri[tid] = tid;
        __syncthreads();
#pragma unroll
        for (int s = 64; s > 0; s >>= 1) {
            if (tid < s && rv[tid + s] > rv[tid]) {
                rv[tid] = rv[tid + s];
                ri[tid] = ri[tid + s];
            }
            __syncthreads();
        }
        if (tid == 0) {
            const int w = ri[0];
            topk_s[row * TOPK + k] = rv[0];
            topk_i[row * TOPK + k] = si[w];
            sv[w] = -INFINITY;
        }
        __syncthreads();
    }
}

// ---------------------------------------------------------------------------
// Phase 2: softmax over 32 + gather-weighted-sum of pool rows.
// One block per query row; 256 threads x float4 = 1024 columns.
// ---------------------------------------------------------------------------
__global__ void __launch_bounds__(256)
softmax_gather_kernel(const float* __restrict__ topk_s,
                      const int*   __restrict__ topk_i,
                      const float* __restrict__ pool,        // [65536][1024] f32
                      unsigned short* __restrict__ agg_bf) { // [4096][1024] bf16
    const int row = blockIdx.x;
    const int tid = threadIdx.x;

    __shared__ float w[TOPK];
    __shared__ int   idx[TOPK];

    if (tid < TOPK) {
        w[tid]   = topk_s[row * TOPK + tid];
        idx[tid] = topk_i[row * TOPK + tid];
    }
    __syncthreads();
    if (tid == 0) {
        float m = w[0];
        for (int j = 1; j < TOPK; ++j) m = fmaxf(m, w[j]);
        float s = 0.f;
        for (int j = 0; j < TOPK; ++j) { const float e = __expf(w[j] - m); w[j] = e; s += e; }
        const float inv = 1.f / s;
        for (int j = 0; j < TOPK; ++j) w[j] *= inv;
    }
    __syncthreads();

    const int d = tid * 4;
    float4 acc = make_float4(0.f, 0.f, 0.f, 0.f);
#pragma unroll 4
    for (int k = 0; k < TOPK; ++k) {
        const float wk = w[k];
        const float4 v = *(const float4*)(pool + (size_t)idx[k] * DP + d);
        acc.x += wk * v.x;
        acc.y += wk * v.y;
        acc.z += wk * v.z;
        acc.w += wk * v.w;
    }
    unsigned short* o = agg_bf + (size_t)row * DP + d;
    o[0] = f32_to_bf16_bits(acc.x);
    o[1] = f32_to_bf16_bits(acc.y);
    o[2] = f32_to_bf16_bits(acc.z);
    o[3] = f32_to_bf16_bits(acc.w);
}

// ---------------------------------------------------------------------------
// Phase 3: out = agg @ W_out^T  (M=4096, N=1024, K=1024), bf16 WMMA f32-acc.
// Grid (256, 16): 16-row x 64-col tiles; block 128 (4 waves, 16 cols each).
// A/B fragments double-buffered across the K loop.
// ---------------------------------------------------------------------------
__global__ void __launch_bounds__(128)
outproj_kernel(const unsigned short* __restrict__ agg_bf,   // [4096][1024] bf16
               const unsigned short* __restrict__ wb,       // [1024][1024] bf16
               float* __restrict__ out) {                   // [4096][1024] f32
    const int tm   = blockIdx.x;          // 0..255
    const int tn   = blockIdx.y;          // 0..15
    const int tid  = threadIdx.x;
    const int wave = tid >> 5;
    const int lane = tid & 31;
    const int lhalf = lane >> 4;
    const int lrow  = lane & 15;

    const unsigned short* arow = agg_bf + (size_t)(tm * 16 + lrow) * DP;
    const unsigned short* brow = wb + (size_t)(tn * 64 + wave * 16 + lrow) * DP;

    v8f c = {0.f, 0.f, 0.f, 0.f, 0.f, 0.f, 0.f, 0.f};
    v16bf a = load_a_frag(arow, 0, lhalf);
    v16bf b = load_b_frag(brow, 0, lhalf);
#pragma unroll
    for (int kk = 0; kk < 32; ++kk) {
        v16bf na = a, nb = b;
        if (kk < 31) {
            na = load_a_frag(arow, (kk + 1) * 32, lhalf);
            nb = load_b_frag(brow, (kk + 1) * 32, lhalf);
        }
        c = __builtin_amdgcn_wmma_f32_16x16x32_bf16(
                false, a, false, b, (short)0, c, false, false);
        a = na;
        b = nb;
    }

    const int gcol = tn * 64 + wave * 16 + lrow;
#pragma unroll
    for (int r = 0; r < 8; ++r) {
        const int grow = tm * 16 + r + 8 * lhalf;
        out[(size_t)grow * DP + gcol] = c[r];
    }
}

// ---------------------------------------------------------------------------
// Host-side launcher
// ---------------------------------------------------------------------------
extern "C" void kernel_launch(void* const* d_in, const int* in_sizes, int n_in,
                              void* d_out, int out_size, void* d_ws, size_t ws_size,
                              hipStream_t stream) {
    (void)in_sizes; (void)n_in; (void)out_size; (void)ws_size;

    const float* query = (const float*)d_in[0];  // [4,1024,512]
    const float* pool  = (const float*)d_in[1];  // [65536,1024]
    const float* keys  = (const float*)d_in[2];  // [65536,512]
    const float* W_out = (const float*)d_in[3];  // [1024,1024]
    float* out = (float*)d_out;                  // [4,1024,1024]

    // Workspace carve-up (256B aligned slabs).
    char* ws = (char*)d_ws;
    size_t off = 0;
    auto carve = [&](size_t bytes) {
        void* p = ws + off;
        off += (bytes + 255) & ~(size_t)255;
        return p;
    };
    unsigned short* qb    = (unsigned short*)carve((size_t)BS_ROWS * DR * 2);  //  4 MB
    unsigned short* kb    = (unsigned short*)carve((size_t)POOL_P * DR * 2);   // 64 MB
    unsigned short* wb    = (unsigned short*)carve((size_t)DP * DP * 2);       //  2 MB
    float* tks_p = (float*)carve((size_t)NSPLIT * BS_ROWS * TOPK * 4);         //  2 MB
    int*   tki_p = (int*)carve((size_t)NSPLIT * BS_ROWS * TOPK * 4);           //  2 MB
    float* tks   = (float*)carve((size_t)BS_ROWS * TOPK * 4);                  // 512 KB
    int*   tki   = (int*)carve((size_t)BS_ROWS * TOPK * 4);                    // 512 KB
    unsigned short* aggb  = (unsigned short*)carve((size_t)BS_ROWS * DP * 2);  //  8 MB

    // Phase 0: bf16 conversions.
    cvt_f32_bf16_kernel<<<1024, 256, 0, stream>>>(query, qb, BS_ROWS * DR);
    cvt_f32_bf16_kernel<<<4096, 256, 0, stream>>>(keys,  kb, POOL_P * DR);
    cvt_f32_bf16_kernel<<<512,  256, 0, stream>>>(W_out, wb, DP * DP);

    // Phase 1: fused scores GEMM + streaming partial top-32 (4 P-splits).
    scores_topk_kernel<<<dim3(BS_ROWS / 32, NSPLIT), 128, 0, stream>>>(
        qb, kb, tks_p, tki_p);

    // Phase 1b: merge partial lists.
    merge_topk_kernel<<<BS_ROWS, NSPLIT * TOPK, 0, stream>>>(tks_p, tki_p, tks, tki);

    // Phase 2: softmax + gather-aggregate.
    softmax_gather_kernel<<<BS_ROWS, 256, 0, stream>>>(tks, tki, pool, aggb);

    // Phase 3: output projection.
    outproj_kernel<<<dim3(BS_ROWS / 16, DP / 64), 128, 0, stream>>>(aggb, wb, out);
}
